// BioContrastiveModel_89936615178813
// MI455X (gfx1250) — matrix operand order repaired and checked by reading, not compiled, not run on hardware
//
#include <hip/hip_runtime.h>
#include <hip/hip_bf16.h>
#include <math.h>

#define N_CELLS 100000
#define R_REG   100
#define RP      112          // R padded to 7 WMMA col-tiles
#define D_DIM   512
#define H_DIM   256
#define LN_EPS  1e-5f
#define TILES_N (N_CELLS / 16)   // 6250 row tiles (exact)

typedef __bf16 bf16_t;
typedef bf16_t v16bf __attribute__((ext_vector_type(16)));
typedef float  v8f   __attribute__((ext_vector_type(8)));

__device__ __forceinline__ unsigned short f2bfu(float f) {
    unsigned u = __float_as_uint(f);
    u = (u + 0x7FFFu + ((u >> 16) & 1u)) >> 16;
    return (unsigned short)u;
}

#if __has_builtin(__builtin_amdgcn_cvt_pk_bf16_f32)
// packed RNE convert: 1 instruction per 2 elements
__device__ __forceinline__ unsigned pack2_bf16(float lo, float hi) {
    auto p = __builtin_amdgcn_cvt_pk_bf16_f32(lo, hi);
    unsigned w;
    __builtin_memcpy(&w, &p, 4);
    return w;
}
#else
// round-to-nearest (half-up) then pack both high halves with one v_perm_b32
__device__ __forceinline__ unsigned pack2_bf16(float lo, float hi) {
    unsigned a = __float_as_uint(lo) + 0x8000u;
    unsigned b = __float_as_uint(hi) + 0x8000u;
    // out.b0 = a.b2, out.b1 = a.b3, out.b2 = b.b2, out.b3 = b.b3
    return __byte_perm(a, b, 0x7632);
}
#endif

__device__ __forceinline__ v16bf make_a(float4 f0, float4 f1, float4 f2, float4 f3) {
    union { unsigned w[8]; v16bf v; } u;
    u.w[0] = pack2_bf16(f0.x, f0.y);
    u.w[1] = pack2_bf16(f0.z, f0.w);
    u.w[2] = pack2_bf16(f1.x, f1.y);
    u.w[3] = pack2_bf16(f1.z, f1.w);
    u.w[4] = pack2_bf16(f2.x, f2.y);
    u.w[5] = pack2_bf16(f2.z, f2.w);
    u.w[6] = pack2_bf16(f3.x, f3.y);
    u.w[7] = pack2_bf16(f3.z, f3.w);
    return u.v;
}

__device__ __forceinline__ float red16_max(float v) {
#pragma unroll
    for (int m = 1; m < 16; m <<= 1) v = fmaxf(v, __shfl_xor(v, m, 32));
    return v;
}
__device__ __forceinline__ float red16_sum(float v) {
#pragma unroll
    for (int m = 1; m < 16; m <<= 1) v += __shfl_xor(v, m, 32);
    return v;
}
__device__ __forceinline__ float gelu_exact(float x) {
    return 0.5f * x * (1.0f + erff(x * 0.70710678118654752f));
}
__device__ __forceinline__ float sigmoidf_(float x) {
    return 1.0f / (1.0f + __expf(-x));
}

// ---------------------------------------------------------------- utilities
__global__ void k_zero_f32(float* __restrict__ p, int n) {
    int i = blockIdx.x * blockDim.x + threadIdx.x;
    if (i < n) p[i] = 0.0f;
}

__global__ void k_conv_protos(const float* __restrict__ P, unsigned short* __restrict__ pb) {
    int idx = blockIdx.x * blockDim.x + threadIdx.x;
    if (idx >= RP * D_DIM) return;
    int r = idx / D_DIM;
    pb[idx] = (r < R_REG) ? f2bfu(P[idx]) : (unsigned short)0;
}

// ua_w is [D][H]; store transposed bf16 [H][D] so B columns are contiguous
__global__ void k_conv_uaw(const float* __restrict__ uaw, unsigned short* __restrict__ ub) {
    int idx = blockIdx.x * blockDim.x + threadIdx.x;
    if (idx >= H_DIM * D_DIM) return;
    int col = idx / D_DIM, k = idx % D_DIM;
    ub[idx] = f2bfu(uaw[k * H_DIM + col]);
}

// ------------------------------------------- K1: sim softmax -> logits[N]
__global__ __launch_bounds__(256) void k_sim_softmax(
    const float* __restrict__ zl, const unsigned short* __restrict__ pb,
    const int* __restrict__ regions, const float* __restrict__ temperature,
    float* __restrict__ logits)
{
    int wave = threadIdx.x >> 5, lane = threadIdx.x & 31;
    int tile = blockIdx.x * 8 + wave;
    if (tile >= TILES_N) return;
    int row0 = tile * 16;
    int h = lane >> 4, n = lane & 15;
    int arow = row0 + n;

    v8f c[7];
#pragma unroll
    for (int t = 0; t < 7; t++)
#pragma unroll
        for (int g = 0; g < 8; g++) c[t][g] = 0.0f;

    const float* abase = zl + (size_t)arow * D_DIM + h * 16;
    for (int k0 = 0; k0 < D_DIM; k0 += 32) {
        const float4* a0p = (const float4*)(abase + k0);
        const float4* a1p = (const float4*)(abase + k0 + 8);
        v16bf a = make_a(a0p[0], a0p[1], a1p[0], a1p[1]);
#pragma unroll
        for (int t = 0; t < 7; t++) {
            const v16bf* bp = (const v16bf*)(pb + ((size_t)(t * 16 + n) * D_DIM + k0 + h * 16));
            c[t] = __builtin_amdgcn_wmma_f32_16x16x32_bf16(false, a, false, *bp,
                                                           (short)0, c[t], false, false);
        }
    }

    float invT = 1.0f / temperature[0];
#pragma unroll
    for (int t = 0; t < 7; t++) {
        bool valid = (t * 16 + n) < R_REG;
#pragma unroll
        for (int g = 0; g < 8; g++) c[t][g] = valid ? c[t][g] * invT : -3.0e38f;
    }

    float myLogit = 0.0f;
#pragma unroll
    for (int g = 0; g < 8; g++) {
        float m = -3.0e38f;
#pragma unroll
        for (int t = 0; t < 7; t++) m = fmaxf(m, c[t][g]);
        m = red16_max(m);
        float ev[7]; float s = 0.0f;
#pragma unroll
        for (int t = 0; t < 7; t++) {
            float e = ((t * 16 + n) < R_REG) ? __expf(c[t][g] - m) : 0.0f;
            ev[t] = e; s += e;
        }
        s = red16_sum(s);
        int row = row0 + g + 8 * h;
        int rg = regions[row];
        int tsel = rg >> 4, nsel = rg & 15;
        float cand = ev[0];
        cand = (tsel == 1) ? ev[1] : cand;
        cand = (tsel == 2) ? ev[2] : cand;
        cand = (tsel == 3) ? ev[3] : cand;
        cand = (tsel == 4) ? ev[4] : cand;
        cand = (tsel == 5) ? ev[5] : cand;
        cand = (tsel == 6) ? ev[6] : cand;
        float v = __shfl(cand, h * 16 + nsel, 32);
        float lg = v / s;
        if (n == g) myLogit = lg;
    }
    if (n < 8) logits[row0 + n + 8 * h] = myLogit;
}

// --------------------------------- K2a / K2b: segment softmax over cells
__global__ void k_segmax(const float* __restrict__ logits, const int* __restrict__ regions,
                         int* __restrict__ segmax_bits) {
    int i = blockIdx.x * blockDim.x + threadIdx.x;
    if (i >= N_CELLS) return;
    // logits are softmax probs in (0,1] -> positive: int compare == float compare
    atomicMax(&segmax_bits[regions[i]], __float_as_int(logits[i]));
}

__global__ void k_exp_denom(const float* __restrict__ logits, const int* __restrict__ regions,
                            const int* __restrict__ segmax_bits, float* __restrict__ evec,
                            float* __restrict__ denom, float* __restrict__ counts) {
    int i = blockIdx.x * blockDim.x + threadIdx.x;
    if (i >= N_CELLS) return;
    int r = regions[i];
    float e = __expf(logits[i] - __int_as_float(segmax_bits[r]));
    evec[i] = e;
    atomicAdd(&denom[r], e);
    atomicAdd(&counts[r], 1.0f);
}

// ---------------- K3: ua = LN(GELU(z @ ua_w + b)); agg[r] += ua * w
__global__ __launch_bounds__(256) void k_ua_agg(
    const float* __restrict__ zl, const unsigned short* __restrict__ uawt,
    const float* __restrict__ ua_b, const float* __restrict__ ua_g,
    const float* __restrict__ ua_bb, const int* __restrict__ regions,
    const float* __restrict__ evec, const float* __restrict__ denom,
    float* __restrict__ agg)
{
    int wave = threadIdx.x >> 5, lane = threadIdx.x & 31;
    int tile = blockIdx.x * 8 + wave;
    if (tile >= TILES_N) return;
    int row0 = tile * 16;
    int h = lane >> 4, n = lane & 15;
    int arow = row0 + n;

    v8f c[16];
#pragma unroll
    for (int t = 0; t < 16; t++)
#pragma unroll
        for (int g = 0; g < 8; g++) c[t][g] = 0.0f;

    const float* abase = zl + (size_t)arow * D_DIM + h * 16;
    for (int k0 = 0; k0 < D_DIM; k0 += 32) {
        const float4* a0p = (const float4*)(abase + k0);
        const float4* a1p = (const float4*)(abase + k0 + 8);
        v16bf a = make_a(a0p[0], a0p[1], a1p[0], a1p[1]);
#pragma unroll
        for (int t = 0; t < 16; t++) {
            const v16bf* bp = (const v16bf*)(uawt + ((size_t)(t * 16 + n) * D_DIM + k0 + h * 16));
            c[t] = __builtin_amdgcn_wmma_f32_16x16x32_bf16(false, a, false, *bp,
                                                           (short)0, c[t], false, false);
        }
    }

#pragma unroll
    for (int t = 0; t < 16; t++) {
        float bias = ua_b[t * 16 + n];
#pragma unroll
        for (int g = 0; g < 8; g++) c[t][g] = gelu_exact(c[t][g] + bias);
    }

#pragma unroll
    for (int g = 0; g < 8; g++) {
        float s1 = 0.0f, s2 = 0.0f;
#pragma unroll
        for (int t = 0; t < 16; t++) { float v = c[t][g]; s1 += v; s2 += v * v; }
        s1 = red16_sum(s1); s2 = red16_sum(s2);
        float mu = s1 * (1.0f / H_DIM);
        float var = s2 * (1.0f / H_DIM) - mu * mu;
        float rstd = rsqrtf(var + LN_EPS);
        int row = row0 + g + 8 * h;
        int rg = regions[row];
        float wi = evec[row] / denom[rg];
#pragma unroll
        for (int t = 0; t < 16; t++) {
            int col = t * 16 + n;
            float y = (c[t][g] - mu) * rstd * ua_g[col] + ua_bb[col];
            atomicAdd(&agg[rg * H_DIM + col], y * wi);
        }
    }
}

// ---- K4: upd = LN(GELU(agg @ up_w + up_b)); z_fused update + residual out
__global__ __launch_bounds__(256) void k_upd(
    const float* __restrict__ agg, const float* __restrict__ up_w,
    const float* __restrict__ up_b, const float* __restrict__ up_g,
    const float* __restrict__ up_bb, const float* __restrict__ counts,
    const float* __restrict__ z_fused, const float* __restrict__ raw_rw,
    float* __restrict__ updf, unsigned short* __restrict__ updbf,
    float* __restrict__ out_zf)
{
    int r = blockIdx.x, t = threadIdx.x;
    __shared__ float sagg[H_DIM];
    __shared__ float sy[D_DIM];
    __shared__ float red[256];
    sagg[t] = agg[r * H_DIM + t];
    __syncthreads();

    float a0 = up_b[t], a1 = up_b[t + 256];
    for (int k = 0; k < H_DIM; k++) {
        float a = sagg[k];
        a0 = fmaf(a, up_w[k * D_DIM + t], a0);
        a1 = fmaf(a, up_w[k * D_DIM + t + 256], a1);
    }
    a0 = gelu_exact(a0); a1 = gelu_exact(a1);
    sy[t] = a0; sy[t + 256] = a1;

    red[t] = a0 + a1;
    for (int s = 128; s > 0; s >>= 1) { __syncthreads(); if (t < s) red[t] += red[t + s]; }
    __syncthreads();
    float total = red[0];
    __syncthreads();
    red[t] = a0 * a0 + a1 * a1;
    for (int s = 128; s > 0; s >>= 1) { __syncthreads(); if (t < s) red[t] += red[t + s]; }
    __syncthreads();
    float totalsq = red[0];

    float mu = total * (1.0f / D_DIM);
    float var = totalsq * (1.0f / D_DIM) - mu * mu;
    float rstd = rsqrtf(var + LN_EPS);
    bool has = counts[r] > 0.0f;
    float rw = sigmoidf_(raw_rw[0]);

    int c0 = t, c1 = t + 256;
    float u0 = (sy[c0] - mu) * rstd * up_g[c0] + up_bb[c0];
    float u1 = (sy[c1] - mu) * rstd * up_g[c1] + up_bb[c1];
    float zf0 = z_fused[r * D_DIM + c0], zf1 = z_fused[r * D_DIM + c1];
    float z0 = has ? u0 : zf0, z1 = has ? u1 : zf1;
    updf[r * D_DIM + c0] = z0;          updf[r * D_DIM + c1] = z1;
    updbf[r * D_DIM + c0] = f2bfu(z0);  updbf[r * D_DIM + c1] = f2bfu(z1);
    out_zf[r * D_DIM + c0] = rw * z0 + (1.0f - rw) * zf0;
    out_zf[r * D_DIM + c1] = rw * z1 + (1.0f - rw) * zf1;
}

// ------- K5: bc = LN(GELU(upd @ db_w1 + b1)) @ db_w2 + b2   (per region)
__global__ __launch_bounds__(256) void k_bc(
    const float* __restrict__ updf, const float* __restrict__ w1,
    const float* __restrict__ b1, const float* __restrict__ g,
    const float* __restrict__ bb, const float* __restrict__ w2,
    const float* __restrict__ b2, float* __restrict__ bcout)
{
    int r = blockIdx.x, t = threadIdx.x;
    __shared__ float srow[D_DIM];
    __shared__ float red[256];
    __shared__ float sh[H_DIM];
    srow[t] = updf[r * D_DIM + t];
    srow[t + 256] = updf[r * D_DIM + t + 256];
    __syncthreads();

    float acc = b1[t];
    for (int k = 0; k < D_DIM; k++) acc = fmaf(srow[k], w1[k * H_DIM + t], acc);
    acc = gelu_exact(acc);

    red[t] = acc;
    for (int s = 128; s > 0; s >>= 1) { __syncthreads(); if (t < s) red[t] += red[t + s]; }
    __syncthreads();
    float total = red[0];
    __syncthreads();
    red[t] = acc * acc;
    for (int s = 128; s > 0; s >>= 1) { __syncthreads(); if (t < s) red[t] += red[t + s]; }
    __syncthreads();
    float totalsq = red[0];
    float mu = total * (1.0f / H_DIM);
    float var = totalsq * (1.0f / H_DIM) - mu * mu;
    float rstd = rsqrtf(var + LN_EPS);
    float hn = (acc - mu) * rstd * g[t] + bb[t];
    __syncthreads();
    sh[t] = hn;
    __syncthreads();

    float o0 = b2[t], o1 = b2[t + 256];
    for (int k = 0; k < H_DIM; k++) {
        float a = sh[k];
        o0 = fmaf(a, w2[k * D_DIM + t], o0);
        o1 = fmaf(a, w2[k * D_DIM + t + 256], o1);
    }
    bcout[r * D_DIM + t] = o0;
    bcout[r * D_DIM + t + 256] = o1;
}

// --- K6: sim2 softmax -> weight; z_local_res = rw*(bc[rg]*w) + (1-rw)*z
__global__ __launch_bounds__(256) void k_down(
    const float* __restrict__ zl, const unsigned short* __restrict__ updbf,
    const int* __restrict__ regions, const float* __restrict__ temperature,
    const float* __restrict__ raw_rw, const float* __restrict__ bc,
    float* __restrict__ out_zl)
{
    int wave = threadIdx.x >> 5, lane = threadIdx.x & 31;
    int tile = blockIdx.x * 8 + wave;
    if (tile >= TILES_N) return;
    int row0 = tile * 16;
    int h = lane >> 4, n = lane & 15;
    int arow = row0 + n;

    v8f c[7];
#pragma unroll
    for (int t = 0; t < 7; t++)
#pragma unroll
        for (int g = 0; g < 8; g++) c[t][g] = 0.0f;

    const float* abase = zl + (size_t)arow * D_DIM + h * 16;
    for (int k0 = 0; k0 < D_DIM; k0 += 32) {
        const float4* a0p = (const float4*)(abase + k0);
        const float4* a1p = (const float4*)(abase + k0 + 8);
        v16bf a = make_a(a0p[0], a0p[1], a1p[0], a1p[1]);
#pragma unroll
        for (int t = 0; t < 7; t++) {
            const v16bf* bp = (const v16bf*)(updbf + ((size_t)(t * 16 + n) * D_DIM + k0 + h * 16));
            c[t] = __builtin_amdgcn_wmma_f32_16x16x32_bf16(false, a, false, *bp,
                                                           (short)0, c[t], false, false);
        }
    }

    float invT = 1.0f / temperature[0];
#pragma unroll
    for (int t = 0; t < 7; t++) {
        bool valid = (t * 16 + n) < R_REG;
#pragma unroll
        for (int g = 0; g < 8; g++) c[t][g] = valid ? c[t][g] * invT : -3.0e38f;
    }

    float wgt[8]; int rgs[8];
#pragma unroll
    for (int g = 0; g < 8; g++) {
        float m = -3.0e38f;
#pragma unroll
        for (int t = 0; t < 7; t++) m = fmaxf(m, c[t][g]);
        m = red16_max(m);
        float ev[7]; float s = 0.0f;
#pragma unroll
        for (int t = 0; t < 7; t++) {
            float e = ((t * 16 + n) < R_REG) ? __expf(c[t][g] - m) : 0.0f;
            ev[t] = e; s += e;
        }
        s = red16_sum(s);
        int row = row0 + g + 8 * h;
        int rg = regions[row];
        int tsel = rg >> 4, nsel = rg & 15;
        float cand = ev[0];
        cand = (tsel == 1) ? ev[1] : cand;
        cand = (tsel == 2) ? ev[2] : cand;
        cand = (tsel == 3) ? ev[3] : cand;
        cand = (tsel == 4) ? ev[4] : cand;
        cand = (tsel == 5) ? ev[5] : cand;
        cand = (tsel == 6) ? ev[6] : cand;
        float v = __shfl(cand, h * 16 + nsel, 32);   // uniform within each half-wave
        wgt[g] = v / s;
        rgs[g] = rg;
    }

    float rw = sigmoidf_(raw_rw[0]);
    float omrw = 1.0f - rw;
#pragma unroll
    for (int g = 0; g < 8; g++) {
        int row = row0 + g + 8 * h;
        const float4* bcr = (const float4*)(bc + (size_t)rgs[g] * D_DIM);
        const float4* zr  = (const float4*)(zl + (size_t)row * D_DIM);
        float4* orow = (float4*)(out_zl + (size_t)row * D_DIM);
        float wt = wgt[g];
#pragma unroll
        for (int it = 0; it < 8; it++) {
            int j = n + 16 * it;
            float4 b4 = bcr[j], z4 = zr[j], o;
            o.x = rw * (b4.x * wt) + omrw * z4.x;
            o.y = rw * (b4.y * wt) + omrw * z4.y;
            o.z = rw * (b4.z * wt) + omrw * z4.z;
            o.w = rw * (b4.w * wt) + omrw * z4.w;
            orow[j] = o;
        }
    }
}

// --------------------------------------------------------------- launcher
extern "C" void kernel_launch(void* const* d_in, const int* in_sizes, int n_in,
                              void* d_out, int out_size, void* d_ws, size_t ws_size,
                              hipStream_t stream)
{
    const float* zl      = (const float*)d_in[0];
    const float* zf      = (const float*)d_in[1];
    const int*   regions = (const int*)d_in[2];
    const float* raw_rw  = (const float*)d_in[3];
    const float* protos  = (const float*)d_in[4];
    const float* temp    = (const float*)d_in[5];
    const float* ua_w    = (const float*)d_in[6];
    const float* ua_b    = (const float*)d_in[7];
    const float* ua_g    = (const float*)d_in[8];
    const float* ua_bb   = (const float*)d_in[9];
    const float* up_w    = (const float*)d_in[10];
    const float* up_b    = (const float*)d_in[11];
    const float* up_g    = (const float*)d_in[12];
    const float* up_bb   = (const float*)d_in[13];
    const float* db_w1   = (const float*)d_in[14];
    const float* db_b1   = (const float*)d_in[15];
    const float* db_g    = (const float*)d_in[16];
    const float* db_bb   = (const float*)d_in[17];
    const float* db_w2   = (const float*)d_in[18];
    const float* db_b2   = (const float*)d_in[19];
    (void)in_sizes; (void)n_in; (void)out_size; (void)ws_size;

    float* out_zl = (float*)d_out;
    float* out_zf = out_zl + (size_t)N_CELLS * D_DIM;

    char* ws = (char*)d_ws;
    size_t off = 0;
    auto take = [&](size_t bytes) -> char* {
        char* p = ws + off;
        off = (off + bytes + 255) & ~(size_t)255;
        return p;
    };
    float* logits            = (float*)take((size_t)N_CELLS * 4);
    float* evec              = (float*)take((size_t)N_CELLS * 4);
    size_t z0 = off;
    int*   segmax            = (int*)  take(512);
    float* denom             = (float*)take(512);
    float* counts            = (float*)take(512);
    float* agg               = (float*)take((size_t)R_REG * H_DIM * 4);
    size_t zcount = (off - z0) / 4;
    unsigned short* protosbf = (unsigned short*)take((size_t)RP * D_DIM * 2);
    unsigned short* uawbf    = (unsigned short*)take((size_t)H_DIM * D_DIM * 2);
    float* updf              = (float*)take((size_t)R_REG * D_DIM * 4);
    unsigned short* updbf    = (unsigned short*)take((size_t)RP * D_DIM * 2);
    float* bc                = (float*)take((size_t)R_REG * D_DIM * 4);

    const int TB = 256;
    // zero the atomic targets + padded bf16 z_fused buffer (every call)
    k_zero_f32<<<(int)((zcount + TB - 1) / TB), TB, 0, stream>>>((float*)(ws + z0), (int)zcount);
    k_zero_f32<<<(RP * D_DIM / 2 + TB - 1) / TB, TB, 0, stream>>>((float*)updbf, RP * D_DIM / 2);
    // weight conversions
    k_conv_protos<<<(RP * D_DIM + TB - 1) / TB, TB, 0, stream>>>(protos, protosbf);
    k_conv_uaw<<<(H_DIM * D_DIM + TB - 1) / TB, TB, 0, stream>>>(ua_w, uawbf);

    int gemmBlocks = (TILES_N + 7) / 8;
    k_sim_softmax<<<gemmBlocks, TB, 0, stream>>>(zl, protosbf, regions, temp, logits);
    k_segmax<<<(N_CELLS + TB - 1) / TB, TB, 0, stream>>>(logits, regions, segmax);
    k_exp_denom<<<(N_CELLS + TB - 1) / TB, TB, 0, stream>>>(logits, regions, segmax, evec, denom, counts);
    k_ua_agg<<<gemmBlocks, TB, 0, stream>>>(zl, uawbf, ua_b, ua_g, ua_bb, regions, evec, denom, agg);
    k_upd<<<R_REG, TB, 0, stream>>>(agg, up_w, up_b, up_g, up_bb, counts, zf, raw_rw, updf, updbf, out_zf);
    k_bc<<<R_REG, TB, 0, stream>>>(updf, db_w1, db_b1, db_g, db_bb, db_w2, db_b2, bc);
    k_down<<<gemmBlocks, TB, 0, stream>>>(zl, updbf, regions, temp, raw_rw, bc, out_zl);
}
